// SelfAttention_34230889349396
// MI455X (gfx1250) — matrix operand order repaired and checked by reading, not compiled, not run on hardware
//
#include <hip/hip_runtime.h>

// ---------------------------------------------------------------------------
// MI455X (gfx1250) self-attention, bf16 WMMA pipeline.
//   x[8192,1024] f32; Wq/Wk/Wv [1024,1024] f32 (stored [in,out]).
//   out = softmax((xWq)(xWk)^T / 1024^5) @ (xWv)   [8192,1024] f32
// ---------------------------------------------------------------------------

typedef __attribute__((ext_vector_type(16))) __bf16        v16bf;
typedef __attribute__((ext_vector_type(2)))  __bf16        v2bf;
typedef __attribute__((ext_vector_type(8)))  float         v8f;
typedef __attribute__((ext_vector_type(4)))  unsigned int  u32x4;
typedef __attribute__((ext_vector_type(4)))  float         f32x4;

#define N_SEQ 8192
#define D_DIM 1024
#define INV_SCALE 8.8817841970012523e-16f   // 1/1024^5 == 2^-50 (faithful to ref)

// round-to-nearest-even f32 -> bf16 bit pattern (scalar fallback)
__device__ __forceinline__ unsigned short f2bf(float f) {
  unsigned int u = __float_as_uint(f);
  u += 0x7fffu + ((u >> 16) & 1u);
  return (unsigned short)(u >> 16);
}

// pack two f32 -> packed bf16x2 (one VALU op if the cvt_pk builtin exists)
__device__ __forceinline__ unsigned int pack2bf(float lo, float hi) {
#if __has_builtin(__builtin_amdgcn_cvt_pk_bf16_f32)
  v2bf p = __builtin_amdgcn_cvt_pk_bf16_f32(lo, hi);
  return __builtin_bit_cast(unsigned int, p);
#else
  return (unsigned)f2bf(lo) | ((unsigned)f2bf(hi) << 16);
#endif
}

union FragU { v16bf v; u32x4 q[2]; };

// A-matrix 16x32 bf16 fragment: lanes 0-15 hold K={0..7,16..23}, lanes 16-31
// hold K={8..15,24..31} (ISA 7.12.2).  p = row base at current K-step.
__device__ __forceinline__ v16bf load_frag_a(const unsigned short* p, int hi) {
  FragU f;
  p += hi * 8;
  f.q[0] = *(const u32x4*)(p);        // K + hi*8 + 0..7
  f.q[1] = *(const u32x4*)(p + 16);   // K + hi*8 + 16..23
  return f.v;
}
// B-matrix 32x16 bf16 fragment: lane = N, lanes 0-15 hold K={0..7,16..23},
// lanes 16-31 hold K={8..15,24..31}.  p = row of K / W^T / V^T (column-major B).
__device__ __forceinline__ v16bf load_frag_b(const unsigned short* p, int hi) {
  FragU f;
  p += hi * 8;
  f.q[0] = *(const u32x4*)(p);
  f.q[1] = *(const u32x4*)(p + 16);
  return f.v;
}

__device__ __forceinline__ v8f wmma_bf16(v16bf a, v16bf b, v8f c) {
  return __builtin_amdgcn_wmma_f32_16x16x32_bf16(false, a, false, b,
                                                 (short)0, c, false, false);
}

// ---------------------------------------------------------------------------
// Stage 0a: f32 -> bf16 bulk convert (8 elems / thread)
// ---------------------------------------------------------------------------
__global__ __launch_bounds__(256) void k_cvt_bf16(const float* __restrict__ in,
                                                  unsigned short* __restrict__ out) {
  int i = (blockIdx.x * 256 + threadIdx.x) * 8;
  f32x4 a = *(const f32x4*)(in + i);
  f32x4 b = *(const f32x4*)(in + i + 4);
  u32x4 r;
  r[0] = pack2bf(a[0], a[1]);
  r[1] = pack2bf(a[2], a[3]);
  r[2] = pack2bf(b[0], b[1]);
  r[3] = pack2bf(b[2], b[3]);
  *(u32x4*)(out + i) = r;
}

// ---------------------------------------------------------------------------
// Stage 0b: W[k][n] f32 -> WT[n][k] bf16 (LDS-tiled transpose, 32x32 tiles)
// ---------------------------------------------------------------------------
__global__ __launch_bounds__(256) void k_transpose_cvt(const float* __restrict__ W,
                                                       unsigned short* __restrict__ WT) {
  __shared__ float tile[32][33];
  const int tx = threadIdx.x & 31;
  const int ty = threadIdx.x >> 5;
  const int n0 = blockIdx.x * 32;
  const int k0 = blockIdx.y * 32;
#pragma unroll
  for (int r = 0; r < 4; ++r)
    tile[ty + r * 8][tx] = W[(size_t)(k0 + ty + r * 8) * D_DIM + n0 + tx];
  __syncthreads();
#pragma unroll
  for (int r = 0; r < 4; ++r)
    WT[(size_t)(n0 + ty + r * 8) * D_DIM + k0 + tx] = f2bf(tile[tx][ty + r * 8]);
}

// ---------------------------------------------------------------------------
// Stage 1: fused QKV projection.  Y = xb @ W  via WT[n][k].
// Block: 256 thr (8 waves, 2Mx4N), tile 128x128, K-step 32 through LDS.
// z==0 -> Q (row major bf16), z==1 -> K (row major), z==2 -> V^T[d][kv].
// ---------------------------------------------------------------------------
#define LSTR 40   // padded LDS row stride (bf16) -> conflict-free b128 frags

__global__ __launch_bounds__(256) void k_qkv_gemm(
    const unsigned short* __restrict__ xb,
    const unsigned short* __restrict__ WqT,
    const unsigned short* __restrict__ WkT,
    const unsigned short* __restrict__ WvT,
    unsigned short* __restrict__ Qb,
    unsigned short* __restrict__ Kb,
    unsigned short* __restrict__ Vt) {
  __shared__ unsigned short Asm[128 * LSTR];
  __shared__ unsigned short Bsm[128 * LSTR];

  const int which = blockIdx.z;
  const unsigned short* WT = (which == 0) ? WqT : (which == 1) ? WkT : WvT;
  const int m0 = blockIdx.y * 128;
  const int n0 = blockIdx.x * 128;
  const int tid = threadIdx.x;
  const int wave = tid >> 5;
  const int lane = tid & 31;
  const int lm = lane & 15;
  const int hi = lane >> 4;
  const int wm = wave & 1;   // 2 waves along M (64 rows each)
  const int wn = wave >> 1;  // 4 waves along N (32 cols each)

  const v8f vzero = {0.f, 0.f, 0.f, 0.f, 0.f, 0.f, 0.f, 0.f};
  v8f acc[4][2];
#pragma unroll
  for (int mt = 0; mt < 4; ++mt)
#pragma unroll
    for (int nt = 0; nt < 2; ++nt) acc[mt][nt] = vzero;

  for (int k0 = 0; k0 < D_DIM; k0 += 32) {
#pragma unroll
    for (int i = 0; i < 2; ++i) {
      int idx = tid + i * 256;
      int r = idx >> 2;
      int c = (idx & 3) * 8;
      *(u32x4*)(&Asm[r * LSTR + c]) =
          *(const u32x4*)(&xb[(size_t)(m0 + r) * D_DIM + k0 + c]);
      *(u32x4*)(&Bsm[r * LSTR + c]) =
          *(const u32x4*)(&WT[(size_t)(n0 + r) * D_DIM + k0 + c]);
    }
    __syncthreads();

    v16bf aF[4], bF[2];
#pragma unroll
    for (int mt = 0; mt < 4; ++mt)
      aF[mt] = load_frag_a(&Asm[(wm * 64 + mt * 16 + lm) * LSTR], hi);
#pragma unroll
    for (int nt = 0; nt < 2; ++nt)
      bF[nt] = load_frag_b(&Bsm[(wn * 32 + nt * 16 + lm) * LSTR], hi);
#pragma unroll
    for (int mt = 0; mt < 4; ++mt)
#pragma unroll
      for (int nt = 0; nt < 2; ++nt)
        acc[mt][nt] = wmma_bf16(aF[mt], bF[nt], acc[mt][nt]);
    __syncthreads();
  }

  // Epilogue.  C layout: lane N = lm, VGPR j -> M = j + 8*hi.
  if (which < 2) {
    unsigned short* Y = (which == 0) ? Qb : Kb;
#pragma unroll
    for (int mt = 0; mt < 4; ++mt)
#pragma unroll
      for (int nt = 0; nt < 2; ++nt) {
        int col = n0 + wn * 32 + nt * 16 + lm;
#pragma unroll
        for (int j = 0; j < 8; ++j) {
          int row = m0 + wm * 64 + mt * 16 + 8 * hi + j;
          Y[(size_t)row * D_DIM + col] = f2bf(acc[mt][nt][j]);
        }
      }
  } else {
    // V^T[d][kv]: lane's 8 values are contiguous along kv -> one b128 store
#pragma unroll
    for (int mt = 0; mt < 4; ++mt)
#pragma unroll
      for (int nt = 0; nt < 2; ++nt) {
        int drow = n0 + wn * 32 + nt * 16 + lm;
        int kcol = m0 + wm * 64 + mt * 16 + 8 * hi;
        u32x4 r;
        r[0] = pack2bf(acc[mt][nt][0], acc[mt][nt][1]);
        r[1] = pack2bf(acc[mt][nt][2], acc[mt][nt][3]);
        r[2] = pack2bf(acc[mt][nt][4], acc[mt][nt][5]);
        r[3] = pack2bf(acc[mt][nt][6], acc[mt][nt][7]);
        *(u32x4*)(&Vt[(size_t)drow * N_SEQ + kcol]) = r;
      }
  }
}

// ---------------------------------------------------------------------------
// Stage 2: flash attention.  Block = 32 q rows, 256 thr (8 waves).
// Per KV step of 128: wave w computes S[0:32, 16w..16w+16) (64 WMMAs over D,
// each K B-fragment reused for both 16-row halves), cooperative online
// softmax in LDS, then wave w accumulates the O d-slice [128w..128w+128)
// for both halves (64 WMMAs, each V^T B-fragment reused twice).
// K and V^T are L2-resident (32 MB << 192 MB L2): B fragments stream from
// global; ~8 GB total L2 traffic vs 274 GFLOP of bf16 WMMA.
// ---------------------------------------------------------------------------
#define QROWS 32
#define QSTR 1032   // 1024 + 8 pad (bf16)
#define SSTR 132    // 128 + 4 pad (f32)
#define PSTR 136    // 128 + 8 pad (bf16)

__global__ __launch_bounds__(256) void k_attention(
    const unsigned short* __restrict__ Qg,
    const unsigned short* __restrict__ Kg,
    const unsigned short* __restrict__ Vt,
    float* __restrict__ out) {
  __shared__ unsigned short Qs[QROWS * QSTR];   // 66 KB
  __shared__ float          Ss[QROWS * SSTR];   // 16.9 KB
  __shared__ unsigned short Ps[QROWS * PSTR];   // 8.7 KB
  __shared__ float pmax[QROWS][8];
  __shared__ float psum[QROWS][8];
  __shared__ float mrow[QROWS], lrow[QROWS], arow[QROWS];

  const int q0 = blockIdx.x * QROWS;
  const int tid = threadIdx.x;
  const int wave = tid >> 5;
  const int lane = tid & 31;
  const int lm = lane & 15;
  const int hi = lane >> 4;

  // load Q tile (32 x 1024 bf16) into LDS, padded stride
#pragma unroll
  for (int i = 0; i < 16; ++i) {
    int idx = tid + i * 256;
    int r = idx >> 7;
    int c = (idx & 127) * 8;
    *(u32x4*)(&Qs[r * QSTR + c]) =
        *(const u32x4*)(&Qg[(size_t)(q0 + r) * D_DIM + c]);
  }
  if (tid < QROWS) { mrow[tid] = -3.0e38f; lrow[tid] = 0.0f; }
  __syncthreads();

  const v8f vzero = {0.f, 0.f, 0.f, 0.f, 0.f, 0.f, 0.f, 0.f};
  v8f Oacc[2][8];
#pragma unroll
  for (int h = 0; h < 2; ++h)
#pragma unroll
    for (int t = 0; t < 8; ++t) Oacc[h][t] = vzero;

  for (int kv0 = 0; kv0 < N_SEQ; kv0 += 128) {
    // ---- S tiles: this wave's 16 kv columns x both q-row halves ----
    const unsigned short* krow = &Kg[(size_t)(kv0 + wave * 16 + lm) * D_DIM];
    if (kv0 + 128 < N_SEQ)   // prefetch next step's K rows into cache
      __builtin_prefetch(krow + 128 * D_DIM, 0, 0);
    v8f s0 = vzero, s1 = vzero;
#pragma unroll 4
    for (int kd = 0; kd < D_DIM; kd += 32) {
      v16bf b  = load_frag_b(krow + kd, hi);
      v16bf a0 = load_frag_a(&Qs[lm * QSTR + kd], hi);
      v16bf a1 = load_frag_a(&Qs[(16 + lm) * QSTR + kd], hi);
      s0 = wmma_bf16(a0, b, s0);
      s1 = wmma_bf16(a1, b, s1);
    }
#pragma unroll
    for (int j = 0; j < 8; ++j) {
      Ss[(j + 8 * hi) * SSTR + wave * 16 + lm]      = s0[j] * INV_SCALE;
      Ss[(16 + j + 8 * hi) * SSTR + wave * 16 + lm] = s1[j] * INV_SCALE;
    }
    __syncthreads();

    // ---- online softmax: partial max (32 rows x 8 chunks of 16) ----
    {
      int r = tid & 31, c = tid >> 5;
      const float* sp = &Ss[r * SSTR + c * 16];
      float mx = sp[0];
#pragma unroll
      for (int i = 1; i < 16; ++i) mx = fmaxf(mx, sp[i]);
      pmax[r][c] = mx;
    }
    __syncthreads();
    if (tid < QROWS) {
      int r = tid;
      float mx = pmax[r][0];
#pragma unroll
      for (int c = 1; c < 8; ++c) mx = fmaxf(mx, pmax[r][c]);
      float mn = fmaxf(mrow[r], mx);
      arow[r] = __expf(mrow[r] - mn);
      mrow[r] = mn;
    }
    __syncthreads();
    // ---- exp, write bf16 P, partial sums ----
    {
      int r = tid & 31, c = tid >> 5;
      float mn = mrow[r];
      const float* sp = &Ss[r * SSTR + c * 16];
      float e[16], sum = 0.f;
#pragma unroll
      for (int i = 0; i < 16; ++i) { e[i] = __expf(sp[i] - mn); sum += e[i]; }
      u32x4 p0, p1;
      p0[0] = pack2bf(e[0], e[1]);   p0[1] = pack2bf(e[2], e[3]);
      p0[2] = pack2bf(e[4], e[5]);   p0[3] = pack2bf(e[6], e[7]);
      p1[0] = pack2bf(e[8], e[9]);   p1[1] = pack2bf(e[10], e[11]);
      p1[2] = pack2bf(e[12], e[13]); p1[3] = pack2bf(e[14], e[15]);
      *(u32x4*)(&Ps[r * PSTR + c * 16])     = p0;
      *(u32x4*)(&Ps[r * PSTR + c * 16 + 8]) = p1;
      psum[r][c] = sum;
    }
    __syncthreads();
    if (tid < QROWS) {
      int r = tid;
      float sm = 0.f;
#pragma unroll
      for (int c = 0; c < 8; ++c) sm += psum[r][c];
      lrow[r] = lrow[r] * arow[r] + sm;
    }

    // ---- rescale O by alpha (per accumulator row) ----
#pragma unroll
    for (int h = 0; h < 2; ++h)
#pragma unroll
      for (int j = 0; j < 8; ++j) {
        float aj = arow[h * 16 + j + 8 * hi];
#pragma unroll
        for (int t = 0; t < 8; ++t) Oacc[h][t][j] *= aj;
      }

    // ---- O += P @ V : this wave's 128-wide d-slice, both halves ----
#pragma unroll
    for (int kk = 0; kk < 128; kk += 32) {
      v16bf a0 = load_frag_a(&Ps[lm * PSTR + kk], hi);
      v16bf a1 = load_frag_a(&Ps[(16 + lm) * PSTR + kk], hi);
#pragma unroll
      for (int nt = 0; nt < 8; ++nt) {
        const unsigned short* vrow =
            &Vt[(size_t)(wave * 128 + nt * 16 + lm) * N_SEQ + kv0 + kk];
        v16bf b = load_frag_b(vrow, hi);
        Oacc[0][nt] = wmma_bf16(a0, b, Oacc[0][nt]);
        Oacc[1][nt] = wmma_bf16(a1, b, Oacc[1][nt]);
      }
    }
    __syncthreads();
  }

  // ---- epilogue: divide by l, store f32 ----
#pragma unroll
  for (int h = 0; h < 2; ++h)
#pragma unroll
    for (int j = 0; j < 8; ++j) {
      float li = 1.0f / lrow[h * 16 + j + 8 * hi];
      int row = q0 + h * 16 + j + 8 * hi;
#pragma unroll
      for (int nt = 0; nt < 8; ++nt) {
        int col = wave * 128 + nt * 16 + lm;
        out[(size_t)row * D_DIM + col] = Oacc[h][nt][j] * li;
      }
    }
}

// ---------------------------------------------------------------------------
// Host-side launcher
// ---------------------------------------------------------------------------
extern "C" void kernel_launch(void* const* d_in, const int* in_sizes, int n_in,
                              void* d_out, int out_size, void* d_ws, size_t ws_size,
                              hipStream_t stream) {
  (void)in_sizes; (void)n_in; (void)out_size; (void)ws_size;
  const float* x  = (const float*)d_in[0];
  const float* Wq = (const float*)d_in[1];
  const float* Wk = (const float*)d_in[2];
  const float* Wv = (const float*)d_in[3];
  float* out = (float*)d_out;

  // workspace layout (bf16 elements), total ~70 MB
  unsigned short* ws = (unsigned short*)d_ws;
  size_t o = 0;
  unsigned short* xb  = ws + o; o += (size_t)N_SEQ * D_DIM;
  unsigned short* WqT = ws + o; o += (size_t)D_DIM * D_DIM;
  unsigned short* WkT = ws + o; o += (size_t)D_DIM * D_DIM;
  unsigned short* WvT = ws + o; o += (size_t)D_DIM * D_DIM;
  unsigned short* Qb  = ws + o; o += (size_t)N_SEQ * D_DIM;
  unsigned short* Kb  = ws + o; o += (size_t)N_SEQ * D_DIM;
  unsigned short* Vt  = ws + o; o += (size_t)D_DIM * N_SEQ;

  k_cvt_bf16<<<(N_SEQ * D_DIM) / (256 * 8), 256, 0, stream>>>(x, xb);
  k_transpose_cvt<<<dim3(32, 32), 256, 0, stream>>>(Wq, WqT);
  k_transpose_cvt<<<dim3(32, 32), 256, 0, stream>>>(Wk, WkT);
  k_transpose_cvt<<<dim3(32, 32), 256, 0, stream>>>(Wv, WvT);
  k_qkv_gemm<<<dim3(D_DIM / 128, N_SEQ / 128, 3), 256, 0, stream>>>(
      xb, WqT, WkT, WvT, Qb, Kb, Vt);
  k_attention<<<N_SEQ / 32, 256, 0, stream>>>(Qb, Kb, Vt, out);
}
